// Seq2Seq_11527692222530
// MI455X (gfx1250) — matrix operand order repaired and checked by reading, not compiled, not run on hardware
//
#include <hip/hip_runtime.h>
#include <hip/hip_bf16.h>

// ---------------- types ----------------
typedef __bf16 bf16x8  __attribute__((ext_vector_type(8)));
typedef __bf16 bf16x16 __attribute__((ext_vector_type(16)));
typedef float  f32x8   __attribute__((ext_vector_type(8)));

// problem constants
#define Bn 128
#define Tn 512
#define Cn 128
#define Hn 512
#define On 128

__device__ __forceinline__ float sigm_(float x) { return 1.0f / (1.0f + __expf(-x)); }
__device__ __forceinline__ float tanh_(float x) { return 1.0f - 2.0f / (__expf(2.0f * x) + 1.0f); }

// ---------------- conversion / setup kernels ----------------

// x[B][T][C] fp32 -> xbf[T][B][C] bf16
__global__ void k_cvt_x(const float* __restrict__ x, __bf16* __restrict__ xbf) {
    int i = blockIdx.x * blockDim.x + threadIdx.x;
    if (i >= Tn * Bn * Cn) return;
    int c = i % Cn;
    int b = (i / Cn) % Bn;
    int t = i / (Cn * Bn);
    xbf[i] = (__bf16)x[(size_t)b * Tn * Cn + (size_t)t * Cn + c];
}

// Build Wcat[4H][Kx+H] bf16 = [Wih | Whh], and bout = bih + bhh
__global__ void k_build_w(const float* __restrict__ wih, const float* __restrict__ whh,
                          const float* __restrict__ bih, const float* __restrict__ bhh,
                          __bf16* __restrict__ wout, float* __restrict__ bout, int Kx) {
    int K = Kx + Hn;
    int total = 4 * Hn * K;
    int i = blockIdx.x * blockDim.x + threadIdx.x;
    if (i < total) {
        int row = i / K, k = i % K;
        float v = (k < Kx) ? wih[(size_t)row * Kx + k] : whh[(size_t)row * Hn + (k - Kx)];
        wout[i] = (__bf16)v;
    }
    if (i < 4 * Hn) bout[i] = bih[i] + bhh[i];
}

// h0/c0 [4][B][H] -> hbuf(parity0) bf16, cbuf fp32
__global__ void k_init_state(const float* __restrict__ h0, const float* __restrict__ c0,
                             __bf16* __restrict__ hbuf0, float* __restrict__ cbuf) {
    int i = blockIdx.x * blockDim.x + threadIdx.x;
    if (i >= 4 * Bn * Hn) return;
    hbuf0[i] = (__bf16)h0[i];
    cbuf[i] = c0[i];
}

// ---------------- fused LSTM cell step (both directions) ----------------
// gates[m][4H] = concat(seg0,seg1,h_in)[m] @ Wcat^T + bias ; then LSTM pointwise.
// One wave = 16 batch rows x 16 hidden columns, 4 accumulators = i/f/g/o gates.
// Block = 8 waves = all 8 m-tiles of ONE h-tile, so all waves in a block issue
// byte-identical W (B-operand) loads -> W is read from L2 once per dir per step,
// the other 7 waves hit the WGP cache. blockIdx.x = h-tile, blockIdx.z = dir.
__global__ void __launch_bounds__(256) k_cell(
    const __bf16* __restrict__ s0f, const __bf16* __restrict__ s0b, int ld0, int len0,
    const __bf16* __restrict__ s1f, const __bf16* __restrict__ s1b, int ld1, int len1,
    const __bf16* __restrict__ hinf, const __bf16* __restrict__ hinb,
    const __bf16* __restrict__ Wf, const __bf16* __restrict__ Wb,
    const float* __restrict__ biasf, const float* __restrict__ biasb,
    float* __restrict__ cstf, float* __restrict__ cstb,
    __bf16* __restrict__ houtf, __bf16* __restrict__ houtb,
    __bf16* __restrict__ seqf, __bf16* __restrict__ seqb, int seqld) {

    const int d = blockIdx.z;
    const __bf16* s0   = d ? s0b   : s0f;
    const __bf16* s1   = d ? s1b   : s1f;
    const __bf16* hin  = d ? hinb  : hinf;
    const __bf16* W    = d ? Wb    : Wf;
    const float*  bias = d ? biasb : biasf;
    float*        cst  = d ? cstb  : cstf;
    __bf16*       hout = d ? houtb : houtf;
    __bf16*       seq  = d ? seqb  : seqf;

    const int lane  = threadIdx.x & 31;
    const int laneR = lane & 15;
    const int laneH = lane >> 4;
    const int m0 = (threadIdx.x >> 5) * 16;  // wave-in-block = m-tile (8 per block)
    const int h0 = blockIdx.x * 16;          // block = h-tile (32 blocks)
    const int K  = len0 + len1 + Hn;

    // A fragment: 16x32 bf16, row = lane&15, 16 contiguous k at half*16
    auto loadA = [&](const __bf16* seg, int ld, int kb) -> bf16x16 {
        const __bf16* p = seg + (size_t)(m0 + laneR) * ld + kb + laneH * 16;
        bf16x8 lo = *(const bf16x8*)p;
        bf16x8 hi = *(const bf16x8*)(p + 8);
        return __builtin_shufflevector(lo, hi, 0, 1, 2, 3, 4, 5, 6, 7,
                                       8, 9, 10, 11, 12, 13, 14, 15);
    };
    // B fragment: column n = W row (g*H + h0 + laneR), 16 contiguous k at half*16
    auto loadB = [&](int g, int kg) -> bf16x16 {
        const __bf16* p = W + (size_t)(g * Hn + h0 + laneR) * K + kg + laneH * 16;
        return *(const bf16x16*)p;
    };

    f32x8 z = {0.f, 0.f, 0.f, 0.f, 0.f, 0.f, 0.f, 0.f};
    f32x8 a0 = z, a1 = z, a2 = z, a3 = z;

#define STEP(SEG, LD, KB, KG)                                                                   \
    {                                                                                           \
        bf16x16 afr = loadA(SEG, LD, KB);                                                       \
        a0 = __builtin_amdgcn_wmma_f32_16x16x32_bf16(false, afr, false, loadB(0, KG), (short)0, \
                                                     a0, false, false);                         \
        a1 = __builtin_amdgcn_wmma_f32_16x16x32_bf16(false, afr, false, loadB(1, KG), (short)0, \
                                                     a1, false, false);                         \
        a2 = __builtin_amdgcn_wmma_f32_16x16x32_bf16(false, afr, false, loadB(2, KG), (short)0, \
                                                     a2, false, false);                         \
        a3 = __builtin_amdgcn_wmma_f32_16x16x32_bf16(false, afr, false, loadB(3, KG), (short)0, \
                                                     a3, false, false);                         \
    }

    int kg = 0;
    for (int kb = 0; kb < len0; kb += 32, kg += 32) STEP(s0, ld0, kb, kg);
    for (int kb = 0; kb < len1; kb += 32, kg += 32) STEP(s1, ld1, kb, kg);
    for (int kb = 0; kb < Hn; kb += 32, kg += 32) STEP(hin, Hn, kb, kg);
#undef STEP

    // LSTM pointwise: this wave owns i/f/g/o for hidden cols [h0, h0+16)
    const int n = h0 + laneR;
    const float bi = bias[0 * Hn + n];
    const float bf = bias[1 * Hn + n];
    const float bg = bias[2 * Hn + n];
    const float bo = bias[3 * Hn + n];
#pragma unroll
    for (int r = 0; r < 8; ++r) {
        const int m = m0 + r + 8 * laneH;
        const size_t idx = (size_t)m * Hn + n;
        const float iv = sigm_(a0[r] + bi);
        const float fv = sigm_(a1[r] + bf);
        const float gv = tanh_(a2[r] + bg);
        const float ov = sigm_(a3[r] + bo);
        const float c2 = fv * cst[idx] + iv * gv;
        cst[idx] = c2;
        const float h2 = ov * tanh_(c2);
        hout[idx] = (__bf16)h2;
        if (seq) seq[(size_t)m * seqld + n] = (__bf16)h2;
    }
}

// ---------------- final linear: pred = [hf|hb] @ lin_w^T + lin_b ----------------
__global__ void k_linear(const __bf16* __restrict__ hf, const __bf16* __restrict__ hb,
                         const float* __restrict__ w, const float* __restrict__ bias,
                         float* __restrict__ out) {
    const int m = blockIdx.y;
    const int o = threadIdx.x; // 128
    float s = bias[o];
    const float* wr = w + (size_t)o * (2 * Hn);
    for (int k = 0; k < Hn; ++k) s += (float)hf[(size_t)m * Hn + k] * wr[k];
    for (int k = 0; k < Hn; ++k) s += (float)hb[(size_t)m * Hn + k] * wr[Hn + k];
    out[(size_t)m * On + o] = s;
}

// ---------------- host ----------------
extern "C" void kernel_launch(void* const* d_in, const int* in_sizes, int n_in,
                              void* d_out, int out_size, void* d_ws, size_t ws_size,
                              hipStream_t stream) {
    const float* x       = (const float*)d_in[0];
    const float* h0      = (const float*)d_in[1];
    const float* c0      = (const float*)d_in[2];
    const float* e_wih0  = (const float*)d_in[3];
    const float* e_wihr  = (const float*)d_in[4];
    const float* e_whh   = (const float*)d_in[5];
    const float* e_bih   = (const float*)d_in[6];
    const float* e_bhh   = (const float*)d_in[7];
    const float* d_wih0  = (const float*)d_in[8];
    const float* d_wihr  = (const float*)d_in[9];
    const float* d_whh   = (const float*)d_in[10];
    const float* d_bih   = (const float*)d_in[11];
    const float* d_bhh   = (const float*)d_in[12];
    const float* lin_w   = (const float*)d_in[13];
    const float* lin_b   = (const float*)d_in[14];

    const int K0 = Cn + Hn;        // 640
    const int K1 = 2 * Hn + Hn;    // 1536
    const size_t BH = (size_t)Bn * Hn;

    char* ws = (char*)d_ws;
    size_t off = 0;
    auto alloc = [&](size_t bytes) -> void* {
        void* p = ws + off;
        off += (bytes + 255) & ~(size_t)255;
        return p;
    };
    __bf16* xbf  = (__bf16*)alloc((size_t)Tn * Bn * Cn * 2);         // [T][B][C]
    __bf16* xs   = (__bf16*)alloc((size_t)Tn * Bn * 2 * Hn * 2);     // enc L0 out [T][B][2H]
    __bf16* wbuf = (__bf16*)alloc((size_t)2 * 2 * (size_t)4 * Hn * (K0 + K1) * 2);
    float*  bbuf = (float*)alloc((size_t)8 * 4 * Hn * 4);
    __bf16* hbuf = (__bf16*)alloc((size_t)2 * 4 * BH * 2);           // [parity][slot][B][H]
    float*  cbuf = (float*)alloc((size_t)4 * BH * 4);                // [slot][B][H]
    (void)ws_size; (void)in_sizes; (void)n_in; (void)out_size;

    // weight/bias pointer tables
    __bf16* Wp[2][2][2];
    float*  Bp[2][2][2];
    {
        size_t wo = 0;
        for (int s = 0; s < 2; s++)
            for (int l = 0; l < 2; l++)
                for (int d = 0; d < 2; d++) {
                    Wp[s][l][d] = wbuf + wo;
                    wo += (size_t)4 * Hn * (l ? K1 : K0);
                    Bp[s][l][d] = bbuf + ((size_t)(s * 2 + l) * 2 + d) * 4 * Hn;
                }
    }

    // --- setup ---
    {
        int tot = Tn * Bn * Cn;
        k_cvt_x<<<(tot + 255) / 256, 256, 0, stream>>>(x, xbf);
    }
    for (int s = 0; s < 2; s++) {
        const float* wih0 = s ? d_wih0 : e_wih0;
        const float* wihr = s ? d_wihr : e_wihr;
        const float* whh  = s ? d_whh  : e_whh;
        const float* bih  = s ? d_bih  : e_bih;
        const float* bhh  = s ? d_bhh  : e_bhh;
        for (int l = 0; l < 2; l++)
            for (int d = 0; d < 2; d++) {
                const float* wih = (l == 0) ? wih0 + (size_t)d * 4 * Hn * Cn
                                            : wihr + (size_t)d * 4 * Hn * (2 * Hn);
                const float* whhp = whh + (size_t)(l * 2 + d) * 4 * Hn * Hn;
                int Kx = (l == 0) ? Cn : 2 * Hn;
                int tot = 4 * Hn * (Kx + Hn);
                k_build_w<<<(tot + 255) / 256, 256, 0, stream>>>(
                    wih, whhp, bih + (size_t)(l * 2 + d) * 4 * Hn,
                    bhh + (size_t)(l * 2 + d) * 4 * Hn, Wp[s][l][d], Bp[s][l][d], Kx);
            }
    }
    {
        int tot = 4 * Bn * Hn;
        k_init_state<<<(tot + 255) / 256, 256, 0, stream>>>(h0, c0, hbuf, cbuf);
    }

    auto hp = [&](int p, int slot) { return hbuf + ((size_t)p * 4 + slot) * BH; };
    const dim3 cgrid(32, 1, 2);  // blockIdx.x = h-tile, blockIdx.z = dir

    // --- encoder layer 0 (fwd + bwd fused per launch) ---
    for (int t = 0; t < Tn; t++) {
        int rp = t & 1, wp = (t + 1) & 1;
        k_cell<<<cgrid, 256, 0, stream>>>(
            xbf + (size_t)t * Bn * Cn, xbf + (size_t)(Tn - 1 - t) * Bn * Cn, Cn, Cn,
            nullptr, nullptr, 0, 0,
            hp(rp, 0), hp(rp, 1),
            Wp[0][0][0], Wp[0][0][1], Bp[0][0][0], Bp[0][0][1],
            cbuf + 0 * BH, cbuf + 1 * BH,
            hp(wp, 0), hp(wp, 1),
            xs + (size_t)t * Bn * 2 * Hn,
            xs + (size_t)(Tn - 1 - t) * Bn * 2 * Hn + Hn, 2 * Hn);
    }
    // --- encoder layer 1 ---
    for (int t = 0; t < Tn; t++) {
        int rp = t & 1, wp = (t + 1) & 1;
        k_cell<<<cgrid, 256, 0, stream>>>(
            xs + (size_t)t * Bn * 2 * Hn, xs + (size_t)(Tn - 1 - t) * Bn * 2 * Hn, 2 * Hn, 2 * Hn,
            nullptr, nullptr, 0, 0,
            hp(rp, 2), hp(rp, 3),
            Wp[0][1][0], Wp[0][1][1], Bp[0][1][0], Bp[0][1][1],
            cbuf + 2 * BH, cbuf + 3 * BH,
            hp(wp, 2), hp(wp, 3),
            nullptr, nullptr, 0);
    }
    // --- decoder: one bidirectional step per t, carrying (h, c) from encoder ---
    for (int t = 0; t < Tn; t++) {
        int rp = t & 1, wp = (t + 1) & 1;
        // layer 0: both dirs see x_t
        k_cell<<<cgrid, 256, 0, stream>>>(
            xbf + (size_t)t * Bn * Cn, xbf + (size_t)t * Bn * Cn, Cn, Cn,
            nullptr, nullptr, 0, 0,
            hp(rp, 0), hp(rp, 1),
            Wp[1][0][0], Wp[1][0][1], Bp[1][0][0], Bp[1][0][1],
            cbuf + 0 * BH, cbuf + 1 * BH,
            hp(wp, 0), hp(wp, 1),
            nullptr, nullptr, 0);
        // layer 1: input = [h_l0f | h_l0b] of this step (parity wp)
        k_cell<<<cgrid, 256, 0, stream>>>(
            hp(wp, 0), hp(wp, 0), Hn, Hn,
            hp(wp, 1), hp(wp, 1), Hn, Hn,
            hp(rp, 2), hp(rp, 3),
            Wp[1][1][0], Wp[1][1][1], Bp[1][1][0], Bp[1][1][1],
            cbuf + 2 * BH, cbuf + 3 * BH,
            hp(wp, 2), hp(wp, 3),
            nullptr, nullptr, 0);
    }
    // --- final linear on last step's top output (final write parity = 0) ---
    k_linear<<<dim3(1, Bn), On, 0, stream>>>(hp(0, 2), hp(0, 3), lin_w, lin_b, (float*)d_out);
}
